// Encoder_84069689852144
// MI455X (gfx1250) — matrix operand order repaired and checked by reading, not compiled, not run on hardware
//
#include <hip/hip_runtime.h>
#include <hip/hip_bf16.h>
#include <math.h>

#define NN 10000   // nodes
#define EE 100000  // edges
#define HH 32      // heads (== wave32 width)

typedef __attribute__((ext_vector_type(16))) __bf16 v16bf;
typedef __attribute__((ext_vector_type(8)))  float  v8f;

__device__ __forceinline__ float eluf(float v) { return v > 0.0f ? v : (expf(v) - 1.0f); }

// order-preserving float<->uint mapping for atomicMax on floats
__device__ __forceinline__ unsigned f2ord(float f) {
  unsigned u = __float_as_uint(f);
  return (u & 0x80000000u) ? ~u : (u | 0x80000000u);
}
__device__ __forceinline__ float ord2f(unsigned u) {
  return (u & 0x80000000u) ? __uint_as_float(u & 0x7FFFFFFFu) : __uint_as_float(~u);
}

// ---------------- conversion / transpose ----------------
__global__ void k_f32_to_bf16(const float* __restrict__ X, __bf16* __restrict__ Y, int n) {
  int i = blockIdx.x * blockDim.x + threadIdx.x;
  if (i < n) Y[i] = (__bf16)X[i];
}

__global__ void k_transpose_bf16(const float* __restrict__ W, __bf16* __restrict__ Wt,
                                 int rows, int cols) {
  int i = blockIdx.x * blockDim.x + threadIdx.x;
  if (i >= rows * cols) return;
  int r = i / cols, c = i - r * cols;
  Wt[(size_t)c * rows + r] = (__bf16)W[i];
}

// ---------------- WMMA GEMM: C[M,Nc] = A[M,K] @ B[K,Nc] ----------------
// A: bf16 row-major.  Bt: bf16 = B transposed, [Nc,K] row-major.
// One wave per 16x16 C tile; K stepped by 32 with v_wmma_f32_16x16x32_bf16.
// A frag (16-bit A 16x32, ISA 7.12.2): lane<16 holds row=lane, K = k0+{0..7,16..23};
// lane>=16 holds row=lane-16, K = k0+{8..15,24..31}.
// B frag: lane<16 holds col=lane, K = k0+{0..7,16..23}; lane>=16 col=lane-16, K = k0+{8..15,24..31}.
// C frag: vgpr r -> row = r + (lane<16?0:8), col = lane&15.
__global__ void k_gemm_bf16_wmma(const __bf16* __restrict__ A, const __bf16* __restrict__ Bt,
                                 void* __restrict__ Cout, int M, int Nc, int K,
                                 int store_bf16, int tiles) {
  int gid  = blockIdx.x * blockDim.x + threadIdx.x;
  int wave = gid >> 5;
  if (wave >= tiles) return;               // wave-uniform: EXEC all-1s inside
  int lane = threadIdx.x & 31;
  int ntn  = Nc >> 4;
  int tm   = wave / ntn;
  int tn   = wave - tm * ntn;
  int half = lane >> 4;
  int l16  = lane & 15;
  int rowA = tm * 16 + l16;
  int colB = tn * 16 + l16;
  v8f acc = {};
  for (int k0 = 0; k0 < K; k0 += 32) {
    const __bf16* ap = A  + (size_t)rowA * K + k0 + half * 8;
    const __bf16* bp = Bt + (size_t)colB * K + k0 + half * 8;
    v16bf a, b;
#pragma unroll
    for (int e = 0; e < 8; ++e)  a[e] = ap[e];        // K = k0+half*8 + 0..7
#pragma unroll
    for (int e = 8; e < 16; ++e) a[e] = ap[8 + e];    // K = k0+half*8 + 16..23
#pragma unroll
    for (int e = 0; e < 8; ++e)  b[e] = bp[e];        // K = k0+half*8 + 0..7
#pragma unroll
    for (int e = 8; e < 16; ++e) b[e] = bp[8 + e];    // K = k0+half*8 + 16..23
    acc = __builtin_amdgcn_wmma_f32_16x16x32_bf16(
        /*neg_a=*/false, a, /*neg_b=*/false, b,
        /*c_mod=*/(short)0, acc, /*reuse_a=*/false, /*reuse_b=*/false);
  }
  int rbase = tm * 16 + half * 8;
  int col   = tn * 16 + l16;
  if (store_bf16) {
    __bf16* C = (__bf16*)Cout;
#pragma unroll
    for (int r = 0; r < 8; ++r) C[(size_t)(rbase + r) * Nc + col] = (__bf16)acc[r];
  } else {
    float* C = (float*)Cout;
#pragma unroll
    for (int r = 0; r < 8; ++r) C[(size_t)(rbase + r) * Nc + col] = acc[r];
  }
}

// ---------------- GAT attention pieces ----------------
// per-(node,head) attention coefficients: aS = <h, a_src>, aD = <h, a_dst>
__global__ void k_attn_coef(const __bf16* __restrict__ Hb, const float* __restrict__ asrc,
                            const float* __restrict__ adst, float* __restrict__ aS,
                            float* __restrict__ aD, int c) {
  int i = blockIdx.x * blockDim.x + threadIdx.x;
  if (i >= NN * HH) return;
  int h = i & (HH - 1);
  int n = i >> 5;
  const __bf16* hp = Hb + ((size_t)n * HH + h) * c;
  const float* s = asrc + h * c;
  const float* d = adst + h * c;
  float s0 = 0.f, s1 = 0.f;
  for (int j = 0; j < c; ++j) { float v = (float)hp[j]; s0 += v * s[j]; s1 += v * d[j]; }
  aS[i] = s0; aD[i] = s1;
}

__global__ void k_edge_max(const int* __restrict__ snd, const int* __restrict__ rcv,
                           const float* __restrict__ aS, const float* __restrict__ aD,
                           unsigned* __restrict__ mmax) {
  int i = blockIdx.x * blockDim.x + threadIdx.x;
  if (i >= EE * HH) return;
  int h = i & (HH - 1), e = i >> 5;
  int s = snd[e], t = rcv[e];
  float lg = aS[s * HH + h] + aD[t * HH + h];
  lg = lg > 0.f ? lg : 0.2f * lg;                        // leaky_relu 0.2
  atomicMax(&mmax[t * HH + h], f2ord(lg));
}

__global__ void k_edge_exp(const int* __restrict__ snd, const int* __restrict__ rcv,
                           const float* __restrict__ aS, const float* __restrict__ aD,
                           const unsigned* __restrict__ mmax,
                           float* __restrict__ ebuf, float* __restrict__ ssum) {
  int i = blockIdx.x * blockDim.x + threadIdx.x;
  if (i >= EE * HH) return;
  int h = i & (HH - 1), e = i >> 5;
  int s = snd[e], t = rcv[e];
  float lg = aS[s * HH + h] + aD[t * HH + h];
  lg = lg > 0.f ? lg : 0.2f * lg;
  float ev = expf(lg - ord2f(mmax[t * HH + h]));
  ebuf[i] = ev;
  atomicAdd(&ssum[t * HH + h], ev);
}

// one wave per edge: lane == head for alpha, lanes own output channels,
// mean-over-heads folded in so the accumulator is only [N, c]
__global__ void k_edge_aggr(const int* __restrict__ snd, const int* __restrict__ rcv,
                            const __bf16* __restrict__ Hb, const float* __restrict__ ebuf,
                            const float* __restrict__ ssum, float* __restrict__ acc, int c) {
  int gid = blockIdx.x * blockDim.x + threadIdx.x;
  int e = gid >> 5;
  if (e >= EE) return;
  int lane = threadIdx.x & 31;
  int s = snd[e], t = rcv[e];
  float ev = ebuf[(size_t)e * HH + lane];
  float ss = ssum[(size_t)t * HH + lane];
  float alpha = ev / (ss + 1e-16f);
  const __bf16* hp = Hb + (size_t)s * HH * c;           // L2-resident bf16 h[src]
  float s0 = 0.f, s1 = 0.f;
  for (int h = 0; h < HH; ++h) {
    float al = __shfl(alpha, h, 32);
    s0 += al * (float)hp[h * c + lane];
    if (c == 64) s1 += al * (float)hp[h * c + lane + 32];
  }
  atomicAdd(&acc[(size_t)t * c + lane], s0);
  if (c == 64) atomicAdd(&acc[(size_t)t * c + lane + 32], s1);
}

// z = elu(acc/H + b); write into d_out slab (stride 96) and optional bf16 copy
__global__ void k_finalize_gat(const float* __restrict__ acc, const float* __restrict__ bias,
                               float* __restrict__ out, int colOff,
                               __bf16* __restrict__ zb, int c) {
  int i = blockIdx.x * blockDim.x + threadIdx.x;
  if (i >= NN * c) return;
  int n = i / c, j = i - n * c;
  float z = eluf(acc[i] * (1.0f / HH) + bias[j]);
  out[(size_t)n * 96 + colOff + j] = z;
  if (zb) zb[i] = (__bf16)z;
}

// ---------------- autoencoder epilogues ----------------
__global__ void k_ae_bn(const float* __restrict__ t1, const float* __restrict__ b,
                        const float* __restrict__ g, const float* __restrict__ be,
                        float* __restrict__ out, int colOff, __bf16* __restrict__ zb, int c) {
  int i = blockIdx.x * blockDim.x + threadIdx.x;
  if (i >= NN * c) return;
  int n = i / c, j = i - n * c;
  float v = eluf(t1[i] + b[j]);
  float y = v * (g[j] * rsqrtf(1.0f + 1e-5f)) + be[j];
  out[(size_t)n * 96 + colOff + j] = y;
  zb[i] = (__bf16)y;
}

__global__ void k_ae_elu_bf(const float* __restrict__ t1, const float* __restrict__ b,
                            __bf16* __restrict__ zb, int c) {
  int i = blockIdx.x * blockDim.x + threadIdx.x;
  if (i >= NN * c) return;
  int j = i % c;
  zb[i] = (__bf16)eluf(t1[i] + b[j]);
}

__global__ void k_ae_sigmoid(const float* __restrict__ t1, const float* __restrict__ b,
                             float* __restrict__ out, int c) {
  int i = blockIdx.x * blockDim.x + threadIdx.x;
  if (i >= NN * c) return;
  int j = i % c;
  float v = t1[i] + b[j];
  out[i] = 1.0f / (1.0f + expf(-v));
}

// ---------------- host-side helpers ----------------
static inline int c256(long n) { return (int)((n + 255) / 256); }

static void gemm(const __bf16* A, const __bf16* Bt, void* C, int M, int Nc, int K,
                 int store_bf16, hipStream_t st) {
  int tiles = (M / 16) * (Nc / 16);
  int blocks = (tiles + 7) / 8;                 // 8 waves / 256-thread block
  k_gemm_bf16_wmma<<<blocks, 256, 0, st>>>(A, Bt, C, M, Nc, K, store_bf16, tiles);
}

static void run_gat_direction(const __bf16* xb,
                              const __bf16* W1t, const float* a1s, const float* a1d, const float* b1,
                              const __bf16* W2t, const float* a2s, const float* a2d, const float* b2,
                              const int* snd, const int* rcv,
                              __bf16* H1b, __bf16* H2b,
                              float* aS, float* aD, unsigned* mmax, float* ssum,
                              float* ebuf, float* acc, __bf16* zb,
                              float* outBase, hipStream_t st) {
  // ---- layer 1: c=64, K=128, Nc=2048 ----
  gemm(xb, W1t, H1b, NN, 2048, 128, /*bf16*/1, st);
  k_attn_coef<<<c256((long)NN * HH), 256, 0, st>>>(H1b, a1s, a1d, aS, aD, 64);
  hipMemsetAsync(mmax, 0, (size_t)NN * HH * 4, st);
  k_edge_max<<<c256((long)EE * HH), 256, 0, st>>>(snd, rcv, aS, aD, mmax);
  hipMemsetAsync(ssum, 0, (size_t)NN * HH * 4, st);
  k_edge_exp<<<c256((long)EE * HH), 256, 0, st>>>(snd, rcv, aS, aD, mmax, ebuf, ssum);
  hipMemsetAsync(acc, 0, (size_t)NN * 64 * 4, st);
  k_edge_aggr<<<c256((long)EE * 32), 256, 0, st>>>(snd, rcv, H1b, ebuf, ssum, acc, 64);
  k_finalize_gat<<<c256((long)NN * 64), 256, 0, st>>>(acc, b1, outBase, 0, zb, 64);
  // ---- layer 2: c=32, K=64, Nc=1024 ----
  gemm(zb, W2t, H2b, NN, 1024, 64, /*bf16*/1, st);
  k_attn_coef<<<c256((long)NN * HH), 256, 0, st>>>(H2b, a2s, a2d, aS, aD, 32);
  hipMemsetAsync(mmax, 0, (size_t)NN * HH * 4, st);
  k_edge_max<<<c256((long)EE * HH), 256, 0, st>>>(snd, rcv, aS, aD, mmax);
  hipMemsetAsync(ssum, 0, (size_t)NN * HH * 4, st);
  k_edge_exp<<<c256((long)EE * HH), 256, 0, st>>>(snd, rcv, aS, aD, mmax, ebuf, ssum);
  hipMemsetAsync(acc, 0, (size_t)NN * 32 * 4, st);
  k_edge_aggr<<<c256((long)EE * 32), 256, 0, st>>>(snd, rcv, H2b, ebuf, ssum, acc, 32);
  k_finalize_gat<<<c256((long)NN * 32), 256, 0, st>>>(acc, b2, outBase, 64, (__bf16*)nullptr, 32);
}

extern "C" void kernel_launch(void* const* d_in, const int* in_sizes, int n_in,
                              void* d_out, int out_size, void* d_ws, size_t ws_size,
                              hipStream_t stream) {
  const float *x;
  const int *ei;
  const float *g1W, *g1as, *g1ad, *g1b, *g2W, *g2as, *g2ad, *g2b;
  const float *o1W, *o1as, *o1ad, *o1b, *o2W, *o2as, *o2ad, *o2b;
  const float *w1, *b1a, *g1a, *be1a, *w2, *b2a, *g2a, *be2a;
  const float *dw1, *db1, *dw2, *db2;

  if (in_sizes[0] == 2 * EE) {
    // jax tree-flatten order (dict keys sorted; 'W' < 'a_dst' < 'a_src' < 'b')
    ei   = (const int*)d_in[0];
    b1a  = (const float*)d_in[1];  b2a  = (const float*)d_in[2];
    be1a = (const float*)d_in[3];  be2a = (const float*)d_in[4];
    db1  = (const float*)d_in[5];  db2  = (const float*)d_in[6];
    dw1  = (const float*)d_in[7];  dw2  = (const float*)d_in[8];
    g1a  = (const float*)d_in[9];  g2a  = (const float*)d_in[10];
    w1   = (const float*)d_in[11]; w2   = (const float*)d_in[12];
    g1W  = (const float*)d_in[13]; g1ad = (const float*)d_in[14];
    g1as = (const float*)d_in[15]; g1b  = (const float*)d_in[16];
    g2W  = (const float*)d_in[17]; g2ad = (const float*)d_in[18];
    g2as = (const float*)d_in[19]; g2b  = (const float*)d_in[20];
    o1W  = (const float*)d_in[21]; o1ad = (const float*)d_in[22];
    o1as = (const float*)d_in[23]; o1b  = (const float*)d_in[24];
    o2W  = (const float*)d_in[25]; o2ad = (const float*)d_in[26];
    o2as = (const float*)d_in[27]; o2b  = (const float*)d_in[28];
    x    = (const float*)d_in[29];
  } else {
    // dict insertion order
    x    = (const float*)d_in[0];  ei   = (const int*)d_in[1];
    g1W  = (const float*)d_in[2];  g1as = (const float*)d_in[3];
    g1ad = (const float*)d_in[4];  g1b  = (const float*)d_in[5];
    g2W  = (const float*)d_in[6];  g2as = (const float*)d_in[7];
    g2ad = (const float*)d_in[8];  g2b  = (const float*)d_in[9];
    o1W  = (const float*)d_in[10]; o1as = (const float*)d_in[11];
    o1ad = (const float*)d_in[12]; o1b  = (const float*)d_in[13];
    o2W  = (const float*)d_in[14]; o2as = (const float*)d_in[15];
    o2ad = (const float*)d_in[16]; o2b  = (const float*)d_in[17];
    w1   = (const float*)d_in[18]; b1a  = (const float*)d_in[19];
    g1a  = (const float*)d_in[20]; be1a = (const float*)d_in[21];
    w2   = (const float*)d_in[22]; b2a  = (const float*)d_in[23];
    g2a  = (const float*)d_in[24]; be2a = (const float*)d_in[25];
    dw1  = (const float*)d_in[26]; db1  = (const float*)d_in[27];
    dw2  = (const float*)d_in[28]; db2  = (const float*)d_in[29];
  }

  const int* src0 = ei;        // edge_index[0]
  const int* dst0 = ei + EE;   // edge_index[1]
  float* out = (float*)d_out;

  // ---------- workspace layout (bump allocator, 256B aligned) ----------
  char* base = (char*)d_ws;
  size_t off = 0;
  auto alloc = [&](size_t bytes) -> void* {
    void* p = base + off;
    off += (bytes + 255) & ~(size_t)255;
    return p;
  };
  __bf16*  xb   = (__bf16*)alloc((size_t)NN * 128 * 2);
  __bf16*  W1ti = (__bf16*)alloc((size_t)2048 * 128 * 2);
  __bf16*  W2ti = (__bf16*)alloc((size_t)1024 * 64 * 2);
  __bf16*  W1to = (__bf16*)alloc((size_t)2048 * 128 * 2);
  __bf16*  W2to = (__bf16*)alloc((size_t)1024 * 64 * 2);
  __bf16*  w1t  = (__bf16*)alloc((size_t)64 * 128 * 2);
  __bf16*  w2t  = (__bf16*)alloc((size_t)32 * 64 * 2);
  __bf16*  dw1t = (__bf16*)alloc((size_t)64 * 32 * 2);
  __bf16*  dw2t = (__bf16*)alloc((size_t)128 * 64 * 2);
  __bf16*  H1b  = (__bf16*)alloc((size_t)NN * 2048 * 2);   // 41 MB, L2-resident
  __bf16*  H2b  = (__bf16*)alloc((size_t)NN * 1024 * 2);   // 20 MB
  float*   aS   = (float*)alloc((size_t)NN * HH * 4);
  float*   aD   = (float*)alloc((size_t)NN * HH * 4);
  unsigned* mmax = (unsigned*)alloc((size_t)NN * HH * 4);
  float*   ssum = (float*)alloc((size_t)NN * HH * 4);
  float*   ebuf = (float*)alloc((size_t)EE * HH * 4);
  float*   acc  = (float*)alloc((size_t)NN * 64 * 4);
  __bf16*  zb   = (__bf16*)alloc((size_t)NN * 64 * 2);
  __bf16*  zb2  = (__bf16*)alloc((size_t)NN * 32 * 2);
  float*   t1   = (float*)alloc((size_t)NN * 128 * 4);

  // ---------- one-time conversions ----------
  k_f32_to_bf16<<<c256((long)NN * 128), 256, 0, stream>>>(x, xb, NN * 128);
  k_transpose_bf16<<<c256(128L * 2048), 256, 0, stream>>>(g1W, W1ti, 128, 2048);
  k_transpose_bf16<<<c256(64L * 1024), 256, 0, stream>>>(g2W, W2ti, 64, 1024);
  k_transpose_bf16<<<c256(128L * 2048), 256, 0, stream>>>(o1W, W1to, 128, 2048);
  k_transpose_bf16<<<c256(64L * 1024), 256, 0, stream>>>(o2W, W2to, 64, 1024);
  k_transpose_bf16<<<c256(128L * 64), 256, 0, stream>>>(w1, w1t, 128, 64);
  k_transpose_bf16<<<c256(64L * 32), 256, 0, stream>>>(w2, w2t, 64, 32);
  k_transpose_bf16<<<c256(32L * 64), 256, 0, stream>>>(dw1, dw1t, 32, 64);
  k_transpose_bf16<<<c256(64L * 128), 256, 0, stream>>>(dw2, dw2t, 64, 128);

  // ---------- x_in: sender = edge[0], receiver = edge[1] ----------
  run_gat_direction(xb, W1ti, g1as, g1ad, g1b, W2ti, g2as, g2ad, g2b,
                    src0, dst0, H1b, H2b, aS, aD, mmax, ssum, ebuf, acc, zb,
                    out + 0, stream);
  // ---------- x_out: roles swapped ----------
  run_gat_direction(xb, W1to, o1as, o1ad, o1b, W2to, o2as, o2ad, o2b,
                    dst0, src0, H1b, H2b, aS, aD, mmax, ssum, ebuf, acc, zb,
                    out + (size_t)NN * 96, stream);

  // ---------- autoencoder ----------
  float* outSelf = out + (size_t)2 * NN * 96;
  float* outRe   = out + (size_t)3 * NN * 96;
  gemm(xb, w1t, t1, NN, 64, 128, 0, stream);
  k_ae_bn<<<c256((long)NN * 64), 256, 0, stream>>>(t1, b1a, g1a, be1a, outSelf, 0, zb, 64);
  gemm(zb, w2t, t1, NN, 32, 64, 0, stream);
  k_ae_bn<<<c256((long)NN * 32), 256, 0, stream>>>(t1, b2a, g2a, be2a, outSelf, 64, zb2, 32);
  gemm(zb2, dw1t, t1, NN, 64, 32, 0, stream);
  k_ae_elu_bf<<<c256((long)NN * 64), 256, 0, stream>>>(t1, db1, zb, 64);
  gemm(zb, dw2t, t1, NN, 128, 64, 0, stream);
  k_ae_sigmoid<<<c256((long)NN * 128), 256, 0, stream>>>(t1, db2, outRe, 128);

  (void)n_in; (void)out_size; (void)ws_size;
}
